// DARQNModel_63780264345847
// MI455X (gfx1250) — matrix operand order repaired and checked by reading, not compiled, not run on hardware
//
#include <hip/hip_runtime.h>
#include <hip/hip_bf16.h>
#include <hip/hip_fp16.h>

// DARQN pipeline for MI455X (gfx1250, wave32, WMMA + TDM):
//  prep:  convert/swizzle weights to f16 WMMA-B layout (one-time, L2-resident)
//  conv1: direct fp32 conv (bandwidth stage), emits f16
//  conv2/conv3: im2col-in-LDS implicit GEMM via v_wmma_f32_16x16x32_f16
//               (A kept row-major in LDS; a lane's A-fragment is two
//                contiguous 16B chunks of one row -> 2x ds_load_b128)
//  recurrent: single persistent 256-thread workgroup, 2048 sequential steps;
//             V tiles double-buffered in LDS and staged one step ahead with
//             tensor_load_to_lds (TDM, overlapped with compute), attention
//             GEMMs in WMMA, LSTM matvec in fp32 VALU.

typedef _Float16 v16h __attribute__((ext_vector_type(16)));
typedef _Float16 v8h  __attribute__((ext_vector_type(8)));
typedef float    v8f  __attribute__((ext_vector_type(8)));
typedef unsigned int u32x4 __attribute__((ext_vector_type(4)));
typedef unsigned int u32x8 __attribute__((ext_vector_type(8)));

// ---- WMMA wave32 B-layout index (16x16x32 f16) ----
// B tile (Kx16): lane = (N&15) + bit3(Kin)*16 ; slot = (Kin&7) + bit4(Kin)*8
__device__ __forceinline__ int b_sw_idx(int k, int n, int NT) {
    int kt = k >> 5, nt = n >> 4, kin = k & 31;
    int lane = (n & 15) + (((kin >> 3) & 1) << 4);
    int slot = (kin & 7) + ((kin >> 4) << 3);
    return ((kt * NT + nt) * 32 + lane) * 16 + slot;
}

// A-fragment for tile row `rowp` (row-major), k0 = kt*32 + (lane>>4)*8:
// halfs K=k0..k0+7 and K=k0+16..k0+23  ->  two 16B LDS loads.
__device__ __forceinline__ v16h load_a_frag(const _Float16* rowp, int k0) {
    v8h lo = *(const v8h*)(rowp + k0);
    v8h hi = *(const v8h*)(rowp + k0 + 16);
    return __builtin_shufflevector(lo, hi, 0, 1, 2, 3, 4, 5, 6, 7,
                                   8, 9, 10, 11, 12, 13, 14, 15);
}

// TDM: async DMA of `nelem` contiguous f16 from global to LDS (one wave issues).
// 2D descriptor: data_size=2B, tensor_dim0=tile_dim0=nelem, dim1=1, type=image.
__device__ __forceinline__ void tdm_load_f16(unsigned lds_off,
                                             const void* src, unsigned nelem) {
    unsigned long long ga = (unsigned long long)(size_t)src;
    u32x4 g0;
    g0[0] = 1u;                                   // count=1 (valid D#)
    g0[1] = lds_off;                              // lds_addr
    g0[2] = (unsigned)ga;                         // global_addr[31:0]
    g0[3] = (unsigned)(ga >> 32) | (2u << 30);    // global_addr[56:32], type=2
    u32x8 g1;
    g1[0] = 1u << 16;                             // wg_mask=0, data_size=2B
    g1[1] = nelem << 16;                          // tensor_dim0[15:0] @bits63:48
    g1[2] = (1u << 16);                           // td0 hi=0; tensor_dim1=1 @bit80
    g1[3] = nelem << 16;                          // tile_dim0 @bits127:112
    g1[4] = 1u;                                   // tile_dim1=1
    g1[5] = nelem;                                // tensor_dim0_stride lo
    g1[6] = 0u;
    g1[7] = 0u;
    asm volatile("tensor_load_to_lds %0, %1" :: "s"(g0), "s"(g1) : "memory");
}

// ---- prep kernels ----
// W is [N][K] row-major fp32; produce B(k,n)=W[n][k] in WMMA-B swizzled f16.
__global__ __launch_bounds__(256) void k_swzB(const float* __restrict__ W,
                                              _Float16* __restrict__ out,
                                              int N, int K) {
    int idx = blockIdx.x * 256 + threadIdx.x;
    if (idx >= N * K) return;
    int n = idx / K, k = idx % K;
    out[b_sw_idx(k, n, N >> 4)] = (_Float16)W[idx];
}

__global__ __launch_bounds__(256) void k_tof16(const float* __restrict__ src,
                                               _Float16* __restrict__ dst, int n) {
    int idx = blockIdx.x * 256 + threadIdx.x;
    if (idx < n) dst[idx] = (_Float16)src[idx];
}

// ---- conv1: [T,1,84,84] -> [T,32,20,20], k=8 s=4, direct fp32 ----
__global__ __launch_bounds__(256) void k_conv1(const float* __restrict__ in,
                                               const float* __restrict__ w,
                                               const float* __restrict__ b,
                                               _Float16* __restrict__ out) {
    __shared__ float wsm[32 * 64];
    __shared__ float bsm[32];
    int tid = threadIdx.x;
    for (int i = tid; i < 32 * 64; i += 256) wsm[i] = w[i];
    if (tid < 32) bsm[tid] = b[tid];
    __syncthreads();
    int f = blockIdx.x;
    int idx = blockIdx.y * 256 + tid;          // 0..12799
    int oc = idx / 400, p = idx % 400;
    int oy = p / 20, ox = p % 20;
    const float* ip = in + f * 7056 + (oy * 4) * 84 + (ox * 4);
    const float* wp = wsm + oc * 64;
    float s = bsm[oc];
#pragma unroll
    for (int ky = 0; ky < 8; ++ky)
#pragma unroll
        for (int kx = 0; kx < 8; ++kx)
            s += ip[ky * 84 + kx] * wp[ky * 8 + kx];
    out[f * 12800 + idx] = (_Float16)fmaxf(s, 0.f);
}

// ---- conv2: implicit GEMM [81(->96) x 512] * [512 x 64], k=4 s=2 ----
__global__ __launch_bounds__(256) void k_conv2(const _Float16* __restrict__ x,
                                               const _Float16* __restrict__ Bsw,
                                               const float* __restrict__ bias,
                                               _Float16* __restrict__ out) {
    __shared__ __align__(16) _Float16 Arow[96 * 512];   // 96KB, row-major im2col
    int tid = threadIdx.x, lane = tid & 31, wv = tid >> 5;
    int f = blockIdx.x;
    const _Float16* xf = x + f * 12800;
    for (int e = tid; e < 96 * 512; e += 256) {
        int m = e >> 9, k = e & 511;
        _Float16 v = (_Float16)0.f;
        if (m < 81) {
            int oy = m / 9, ox = m % 9;
            int ic = k >> 4, r = k & 15, ky = r >> 2, kx = r & 3;
            v = xf[ic * 400 + (oy * 2 + ky) * 20 + (ox * 2 + kx)];
        }
        Arow[e] = v;                                    // coalesced
    }
    __syncthreads();
    int g8 = (lane >> 4) * 8;
    for (int t = wv * 3; t < wv * 3 + 3; ++t) {         // 24 tiles / 8 waves
        int mt = t >> 2, nt = t & 3;
        const _Float16* rowp = &Arow[(mt * 16 + (lane & 15)) * 512];
        v8f acc = {};
        for (int kt = 0; kt < 16; ++kt) {
            v16h a = load_a_frag(rowp, kt * 32 + g8);
            v16h b = *(const v16h*)&Bsw[((kt * 4 + nt) * 32 + lane) * 16];
            acc = __builtin_amdgcn_wmma_f32_16x16x32_f16(false, a, false, b,
                                                         (short)0, acc, false, false);
        }
        int n = nt * 16 + (lane & 15), g = lane >> 4;
        float bn = bias[n];
#pragma unroll
        for (int r = 0; r < 8; ++r) {
            int m = mt * 16 + r + 8 * g;
            if (m < 81) out[f * 5184 + n * 81 + m] = (_Float16)fmaxf(acc[r] + bn, 0.f);
        }
    }
}

// ---- conv3: implicit GEMM [49(->64) x 576] * [576 x 256], k=3 s=1 ----
// output directly as vecs[t][m][ch] (the build_vector transpose)
__global__ __launch_bounds__(256) void k_conv3(const _Float16* __restrict__ x,
                                               const _Float16* __restrict__ Bsw,
                                               const float* __restrict__ bias,
                                               _Float16* __restrict__ out) {
    __shared__ __align__(16) _Float16 Arow[64 * 576];   // 72KB, row-major im2col
    int tid = threadIdx.x, lane = tid & 31, wv = tid >> 5;
    int f = blockIdx.x;
    const _Float16* xf = x + f * 5184;
    for (int e = tid; e < 64 * 576; e += 256) {
        int m = e / 576, k = e % 576;
        _Float16 v = (_Float16)0.f;
        if (m < 49) {
            int oy = m / 7, ox = m % 7;
            int ic = k / 9, r = k % 9, ky = r / 3, kx = r % 3;
            v = xf[ic * 81 + (oy + ky) * 9 + (ox + kx)];
        }
        Arow[e] = v;
    }
    __syncthreads();
    int g8 = (lane >> 4) * 8;
    for (int t = wv * 8; t < wv * 8 + 8; ++t) {         // 64 tiles / 8 waves
        int mt = t >> 4, nt = t & 15;
        const _Float16* rowp = &Arow[(mt * 16 + (lane & 15)) * 576];
        v8f acc = {};
        for (int kt = 0; kt < 18; ++kt) {
            v16h a = load_a_frag(rowp, kt * 32 + g8);
            v16h b = *(const v16h*)&Bsw[((kt * 16 + nt) * 32 + lane) * 16];
            acc = __builtin_amdgcn_wmma_f32_16x16x32_f16(false, a, false, b,
                                                         (short)0, acc, false, false);
        }
        int n = nt * 16 + (lane & 15), g = lane >> 4;
        float bn = bias[n];
#pragma unroll
        for (int r = 0; r < 8; ++r) {
            int m = mt * 16 + r + 8 * g;
            if (m < 49) out[f * 12544 + m * 256 + n] = (_Float16)fmaxf(acc[r] + bn, 0.f);
        }
    }
}

// ---- recurrent: one persistent workgroup, 2048 sequential steps ----
__global__ __launch_bounds__(256) void k_recurrent(
    const _Float16* __restrict__ vecs,              // [2048][49][256]
    const _Float16* __restrict__ B1,                // attn_w1^T swizzled
    const _Float16* __restrict__ B2,                // attn_w2^T swizzled
    const _Float16* __restrict__ Wih,               // [1024][256] f16
    const _Float16* __restrict__ Whh,               // [1024][256] f16
    const float* __restrict__ b1, const float* __restrict__ b2,
    const float* __restrict__ bih, const float* __restrict__ bhh,
    const float* __restrict__ qw, const float* __restrict__ qb,
    float* __restrict__ out) {
    __shared__ __align__(16) _Float16 Vrow[2][64 * 256];  // 64KB: V double-buffer
    __shared__ __align__(16) _Float16 Srow[64 * 256];     // 32KB: s row-major
    __shared__ float logit[49 * 256];                     // ~49KB
    __shared__ float ctxS[256], hS[256], cS[256];
    int tid = threadIdx.x, lane = tid & 31, wv = tid >> 5;
    hS[tid] = 0.f;
    cS[tid] = 0.f;
    // zero padding rows 49..63 once (TDM only writes rows 0..48)
    for (int e = 49 * 256 + tid; e < 64 * 256; e += 256) {
        Vrow[0][e] = (_Float16)0.f;
        Vrow[1][e] = (_Float16)0.f;
    }
    unsigned vlds[2];
    vlds[0] = (unsigned)(size_t)(void*)&Vrow[0][0];       // LDS byte offsets
    vlds[1] = (unsigned)(size_t)(void*)&Vrow[1][0];
    // prologue: stage V(0) into buffer 0
    if (wv == 0) tdm_load_f16(vlds[0], vecs, 12544u);
    int g8 = (lane >> 4) * 8;
    for (int t = 0; t < 2048; ++t) {
        __builtin_amdgcn_s_wait_tensorcnt(0);   // V(t) resident (issuing wave)
        __syncthreads();                        // publish to all waves
        const _Float16* Vcur = &Vrow[t & 1][0];
        // overlap: DMA V(t+1) into the other buffer while computing step t
        if (t + 1 < 2048) {
            if (wv == 0)
                tdm_load_f16(vlds[(t + 1) & 1],
                             vecs + (size_t)(t + 1) * 12544, 12544u);
            __builtin_prefetch(vecs + (size_t)(t + 1) * 12544, 0, 1);
        }
        // ---- GEMM1: s = tanh(V @ W1^T + b1 + h) -> Srow ----
        for (int ti = wv * 8; ti < wv * 8 + 8; ++ti) {
            int mt = ti >> 4, nt = ti & 15;
            const _Float16* rowp = &Vcur[(mt * 16 + (lane & 15)) * 256];
            v8f acc = {};
#pragma unroll
            for (int kt = 0; kt < 8; ++kt) {
                v16h a = load_a_frag(rowp, kt * 32 + g8);
                v16h b = *(const v16h*)&B1[((kt * 16 + nt) * 32 + lane) * 16];
                acc = __builtin_amdgcn_wmma_f32_16x16x32_f16(false, a, false, b,
                                                             (short)0, acc, false, false);
            }
            int n = nt * 16 + (lane & 15), g = lane >> 4;
            float add = b1[n] + hS[n];
#pragma unroll
            for (int r = 0; r < 8; ++r) {
                int m = mt * 16 + r + 8 * g;
                Srow[m * 256 + n] = (_Float16)tanhf(acc[r] + add);
            }
        }
        __syncthreads();
        // ---- GEMM2: logits = s @ W2^T + b2 ----
        for (int ti = wv * 8; ti < wv * 8 + 8; ++ti) {
            int mt = ti >> 4, nt = ti & 15;
            const _Float16* rowp = &Srow[(mt * 16 + (lane & 15)) * 256];
            v8f acc = {};
#pragma unroll
            for (int kt = 0; kt < 8; ++kt) {
                v16h a = load_a_frag(rowp, kt * 32 + g8);
                v16h b = *(const v16h*)&B2[((kt * 16 + nt) * 32 + lane) * 16];
                acc = __builtin_amdgcn_wmma_f32_16x16x32_f16(false, a, false, b,
                                                             (short)0, acc, false, false);
            }
            int n = nt * 16 + (lane & 15), g = lane >> 4;
            float bn = b2[n];
#pragma unroll
            for (int r = 0; r < 8; ++r) {
                int m = mt * 16 + r + 8 * g;
                if (m < 49) logit[m * 256 + n] = acc[r] + bn;
            }
        }
        __syncthreads();
        // ---- softmax over the 256 feature dim, one row per wave ----
        for (int m = wv; m < 49; m += 8) {
            float mx = -3.4e38f;
#pragma unroll
            for (int j = 0; j < 8; ++j) mx = fmaxf(mx, logit[m * 256 + lane + 32 * j]);
#pragma unroll
            for (int off = 16; off; off >>= 1) mx = fmaxf(mx, __shfl_xor(mx, off, 32));
            float ev[8], sm = 0.f;
#pragma unroll
            for (int j = 0; j < 8; ++j) {
                ev[j] = expf(logit[m * 256 + lane + 32 * j] - mx);
                sm += ev[j];
            }
#pragma unroll
            for (int off = 16; off; off >>= 1) sm += __shfl_xor(sm, off, 32);
            float inv = 1.f / sm;
#pragma unroll
            for (int j = 0; j < 8; ++j) logit[m * 256 + lane + 32 * j] = ev[j] * inv;
        }
        __syncthreads();
        // ---- ctx[n] = sum_m w[m,n] * V[m,n]  (V from LDS) ----
        {
            float s = 0.f;
            for (int m = 0; m < 49; ++m)
                s += logit[m * 256 + tid] * (float)Vcur[m * 256 + tid];
            ctxS[tid] = s;
        }
        __syncthreads();
        // ---- LSTM cell (gate order i,f,g,o); thread tid owns unit tid ----
        float gi = bih[tid] + bhh[tid];
        float gf = bih[256 + tid] + bhh[256 + tid];
        float gg = bih[512 + tid] + bhh[512 + tid];
        float go = bih[768 + tid] + bhh[768 + tid];
        const _Float16* wi0 = Wih + tid * 256;
        const _Float16* wh0 = Whh + tid * 256;
#pragma unroll 4
        for (int k = 0; k < 256; ++k) {
            float cv = ctxS[k], hv = hS[k];
            gi += cv * (float)wi0[k]          + hv * (float)wh0[k];
            gf += cv * (float)wi0[65536 + k]  + hv * (float)wh0[65536 + k];
            gg += cv * (float)wi0[131072 + k] + hv * (float)wh0[131072 + k];
            go += cv * (float)wi0[196608 + k] + hv * (float)wh0[196608 + k];
        }
        __syncthreads();   // all hS reads complete before overwrite
        float ii = 1.f / (1.f + expf(-gi));
        float ff = 1.f / (1.f + expf(-gf));
        float oo = 1.f / (1.f + expf(-go));
        float cn = ff * cS[tid] + ii * tanhf(gg);
        hS[tid] = oo * tanhf(cn);
        cS[tid] = cn;
        __syncthreads();
    }
    if (tid < 18) {
        float q = qb[tid];
        for (int k = 0; k < 256; ++k) q += hS[k] * qw[tid * 256 + k];
        out[tid] = q;
    }
}

extern "C" void kernel_launch(void* const* d_in, const int* in_sizes, int n_in,
                              void* d_out, int out_size, void* d_ws, size_t ws_size,
                              hipStream_t stream) {
    (void)in_sizes; (void)n_in; (void)out_size; (void)ws_size;
    const float* frames = (const float*)d_in[0];
    const float* c1w = (const float*)d_in[1];
    const float* c1b = (const float*)d_in[2];
    const float* c2w = (const float*)d_in[3];
    const float* c2b = (const float*)d_in[4];
    const float* c3w = (const float*)d_in[5];
    const float* c3b = (const float*)d_in[6];
    const float* aw1 = (const float*)d_in[7];
    const float* ab1 = (const float*)d_in[8];
    const float* aw2 = (const float*)d_in[9];
    const float* ab2 = (const float*)d_in[10];
    const float* wih = (const float*)d_in[11];
    const float* whh = (const float*)d_in[12];
    const float* bih = (const float*)d_in[13];
    const float* bhh = (const float*)d_in[14];
    const float* qw  = (const float*)d_in[15];
    const float* qb  = (const float*)d_in[16];

    char* ws = (char*)d_ws;
    size_t off = 0;
    auto take = [&](size_t bytes) -> char* {
        char* p = ws + off;
        off = (off + bytes + 255) & ~(size_t)255;
        return p;
    };
    _Float16* c1o  = (_Float16*)take(2048ull * 12800 * 2);   // conv1 out f16
    _Float16* c2o  = (_Float16*)take(2048ull * 5184 * 2);    // conv2 out f16
    _Float16* vecs = (_Float16*)take(2048ull * 12544 * 2);   // conv3 out (vecs) f16
    _Float16* w2s  = (_Float16*)take(64ull * 512 * 2);       // conv2 W, B-swizzled
    _Float16* w3s  = (_Float16*)take(256ull * 576 * 2);      // conv3 W, B-swizzled
    _Float16* a1s  = (_Float16*)take(256ull * 256 * 2);      // attn_w1, B-swizzled
    _Float16* a2s  = (_Float16*)take(256ull * 256 * 2);      // attn_w2, B-swizzled
    _Float16* wihf = (_Float16*)take(1024ull * 256 * 2);     // lstm w_ih f16
    _Float16* whhf = (_Float16*)take(1024ull * 256 * 2);     // lstm w_hh f16

    // weight prep (one-time; ~1.6MB f16, stays L2-resident afterwards)
    k_swzB<<<128, 256, 0, stream>>>(c2w, w2s, 64, 512);
    k_swzB<<<576, 256, 0, stream>>>(c3w, w3s, 256, 576);
    k_swzB<<<256, 256, 0, stream>>>(aw1, a1s, 256, 256);
    k_swzB<<<256, 256, 0, stream>>>(aw2, a2s, 256, 256);
    k_tof16<<<1024, 256, 0, stream>>>(wih, wihf, 262144);
    k_tof16<<<1024, 256, 0, stream>>>(whh, whhf, 262144);

    // batch-parallel CNN
    k_conv1<<<dim3(2048, 50), 256, 0, stream>>>(frames, c1w, c1b, c1o);
    k_conv2<<<2048, 256, 0, stream>>>(c1o, w2s, c2b, c2o);
    k_conv3<<<2048, 256, 0, stream>>>(c2o, w3s, c3b, vecs);

    // sequential attention+LSTM scan: one persistent workgroup
    k_recurrent<<<1, 256, 0, stream>>>(vecs, a1s, a2s, wihf, whhf,
                                       ab1, ab2, bih, bhh, qw, qb, (float*)d_out);
}
